// Encoder_24197845746053
// MI455X (gfx1250) — matrix operand order repaired and checked by reading, not compiled, not run on hardware
//
#include <hip/hip_runtime.h>
#include <hip/hip_bf16.h>
#include <cstdint>
#include <cstddef>

// ---------------------------------------------------------------------------
// Types / constants
// ---------------------------------------------------------------------------
typedef __bf16 bf16_t;
typedef __attribute__((ext_vector_type(16))) __bf16 v16bf;
typedef __attribute__((ext_vector_type(8)))  __bf16 v8bf;
typedef __attribute__((ext_vector_type(4)))  __bf16 v4bf;
typedef __attribute__((ext_vector_type(8)))  float  v8f;

#define DEV __device__ __forceinline__

static constexpr int HID   = 1024;
static constexpr int NH    = 8;
static constexpr int HD    = 128;
static constexpr int LW    = 64;
static constexpr int LIMG  = 4096;
static constexpr int LTOT  = 4160;   // LW + LIMG

DEV bf16_t f2bf(float f) {
  unsigned u = __builtin_bit_cast(unsigned, f);
  u += 0x7FFFu + ((u >> 16) & 1u);              // round-to-nearest-even
  unsigned short h = (unsigned short)(u >> 16);
  return __builtin_bit_cast(bf16_t, h);
}

DEV v8f wmma_bf16(v16bf a, v16bf b, v8f c) {
  // D = A(16x32) * B(32x16) + C, f32 accumulate
  return __builtin_amdgcn_wmma_f32_16x16x32_bf16(false, a, false, b, (short)0, c,
                                                 false, false);
}

DEV v16bf cat8(v8bf a, v8bf b) {
  return __builtin_shufflevector(a, b, 0,1,2,3,4,5,6,7,8,9,10,11,12,13,14,15);
}

DEV v4bf cvt4(float4 f) {
  v4bf b;
  b[0] = f2bf(f.x); b[1] = f2bf(f.y); b[2] = f2bf(f.z); b[3] = f2bf(f.w);
  return b;
}

// ---------------------------------------------------------------------------
// Weight pre-pass: Wt[n][k] = bf16(W[k][n]).  Tiled LDS transpose,
// coalesced reads (over n) and writes (over k).  grid=(N/32, K/32), block=(32,8)
// ---------------------------------------------------------------------------
__global__ void convert_transpose_kernel(const float* __restrict__ W,
                                         bf16_t* __restrict__ Wt, int K, int N) {
  __shared__ float tile[32][33];
  const int kb = blockIdx.y * 32, nb = blockIdx.x * 32;
  for (int i = threadIdx.y; i < 32; i += 8)
    tile[i][threadIdx.x] = W[(size_t)(kb + i) * N + nb + threadIdx.x];
  __syncthreads();
  for (int i = threadIdx.y; i < 32; i += 8)
    Wt[(size_t)(nb + i) * K + kb + threadIdx.x] = f2bf(tile[threadIdx.x][i]);
}

// ---------------------------------------------------------------------------
// WMMA GEMM:  C[M,N] = A[M,K](f32) * W[K,N] (+bias) (+=C if accumulate)
// Weights pre-converted/transposed: Wt[n][k] bf16, row length ldwt (>= K).
// Block = 256 threads (8 waves); block tile 64(M) x 256(N); wave tile 32x64.
// REQUIRES: M % 64 == 0, N % 256 == 0, K % 32 == 0 (true for all call sites).
// A staged through LDS (f32 -> bf16, straight-line vectorized, hoisted
// pointers); B loaded straight from global to registers (32B contiguous per
// lane; the whole weight set is L2-resident on MI455X's 192MB L2).
// ---------------------------------------------------------------------------
__global__ void gemm_wmma_kernel(const float* __restrict__ A, int lda,
                                 const bf16_t* __restrict__ Wt, int ldwt,
                                 const float* __restrict__ bias,
                                 float* __restrict__ C, int ldc,
                                 int M, int N, int K, int accumulate) {
  __shared__ bf16_t As[64][32];   // [m][k]

  const int lane = threadIdx.x & 31;
  const int wave = threadIdx.x >> 5;          // 0..7
  const int hi   = lane >> 4;                 // 0/1
  const int lo16 = lane & 15;
  const int wm0  = (wave >> 2) * 32;          // wave M offset in 64-row tile
  const int wn0  = (wave & 3) * 64;           // wave N offset in 256-col tile
  const int m0   = blockIdx.y * 64;
  const int n0   = blockIdx.x * 256;
  (void)M; (void)N;

  // ---- hoisted staging pointers: thread owns rows (arow, arow+32), quad akq
  const int arow = threadIdx.x >> 3;          // 0..31
  const int akq  = (threadIdx.x & 7) * 4;     // 0,4,...,28
  const float* ap0 = A + (size_t)(m0 + arow) * lda + akq;
  const float* ap1 = A + (size_t)(m0 + arow + 32) * lda + akq;
  bf16_t* as0 = &As[arow][akq];
  bf16_t* as1 = &As[arow + 32][akq];

  // ---- hoisted fragment addresses
  const bf16_t* arow0 = &As[wm0 + lo16][0];        // sm = 0
  const bf16_t* arow1 = &As[wm0 + 16 + lo16][0];   // sm = 1
  const bf16_t* bptr[4];
  for (int sn = 0; sn < 4; ++sn)
    bptr[sn] = Wt + (size_t)(n0 + wn0 + sn * 16 + lo16) * ldwt + hi * 16;

  v8f acc[2][4];
  for (int i = 0; i < 2; ++i)
    for (int j = 0; j < 4; ++j)
      acc[i][j] = v8f{0.f,0.f,0.f,0.f,0.f,0.f,0.f,0.f};

  for (int kb = 0; kb < K; kb += 32) {
    // stage A tile (64 rows x 32 k): 2 x (b128 load -> 4xbf16 -> b64 store)
    *(v4bf*)as0 = cvt4(*(const float4*)(ap0 + kb));
    *(v4bf*)as1 = cvt4(*(const float4*)(ap1 + kb));
    __syncthreads();

    // A fragments from LDS
    v16bf afr[2];
    {
      v8bf a0 = *(const v8bf*)(arow0 + hi * 16);
      v8bf a1 = *(const v8bf*)(arow0 + hi * 16 + 8);
      afr[0] = cat8(a0, a1);
      v8bf b0 = *(const v8bf*)(arow1 + hi * 16);
      v8bf b1 = *(const v8bf*)(arow1 + hi * 16 + 8);
      afr[1] = cat8(b0, b1);
    }
    // B fragments straight from global (bf16, n-major rows: 32B contiguous)
    for (int sn = 0; sn < 4; ++sn) {
      v16bf bfr = *(const v16bf*)(bptr[sn] + kb);
      if (kb + 32 < K) __builtin_prefetch(bptr[sn] + kb + 32, 0, 1);
      for (int sm = 0; sm < 2; ++sm)
        acc[sm][sn] = wmma_bf16(afr[sm], bfr, acc[sm][sn]);
    }
    __syncthreads();
  }

  // epilogue (C layout: vgpr r -> row r + 8*hi; col = lo16)
  for (int sm = 0; sm < 2; ++sm)
    for (int sn = 0; sn < 4; ++sn)
      for (int r = 0; r < 8; ++r) {
        int m = m0 + wm0 + sm * 16 + r + 8 * hi;
        int n = n0 + wn0 + sn * 16 + lo16;
        float v = acc[sm][sn][r];
        if (bias) v += bias[n];
        if (accumulate) v += C[(size_t)m * ldc + n];
        C[(size_t)m * ldc + n] = v;
      }
}

// ---------------------------------------------------------------------------
// Flash attention (one wave per (head, 16-row q tile)).
// Q,K: bf16 [NH][LTOT][HD] ; Vt: bf16 [NH][HD][LTOT] (d-major)
// O:  f32 [LTOT][NH*HD]    (heads concatenated)
// ---------------------------------------------------------------------------
__global__ void flash_attn_kernel(const bf16_t* __restrict__ Q,
                                  const bf16_t* __restrict__ K,
                                  const bf16_t* __restrict__ Vt,
                                  float* __restrict__ O) {
  const int lane = threadIdx.x;
  const int hi   = lane >> 4;
  const int lo16 = lane & 15;
  const int h    = blockIdx.y;
  const int qt   = blockIdx.x;    // 0..LTOT/16-1

  const bf16_t* Qh = Q  + (size_t)h * LTOT * HD;
  const bf16_t* Kh = K  + (size_t)h * LTOT * HD;
  const bf16_t* Vh = Vt + (size_t)h * HD * LTOT;

  // Q A-fragments: 4 chunks of 32 along d
  v16bf qf[4];
  {
    const bf16_t* qrow = Qh + (size_t)(qt * 16 + lo16) * HD;
    for (int c = 0; c < 4; ++c) {
      v8bf a0 = *(const v8bf*)(qrow + c * 32 + hi * 16);
      v8bf a1 = *(const v8bf*)(qrow + c * 32 + hi * 16 + 8);
      qf[c] = cat8(a0, a1);
    }
  }

  float mrow[8], lrow[8];
  for (int r = 0; r < 8; ++r) { mrow[r] = -3.0e38f; lrow[r] = 0.f; }
  v8f o[8];
  for (int dc = 0; dc < 8; ++dc) o[dc] = v8f{0.f,0.f,0.f,0.f,0.f,0.f,0.f,0.f};

  __shared__ bf16_t Pls[16][32];
  const float scale = 0.08838834764831845f;  // 1/sqrt(128)

  for (int kt = 0; kt < LTOT / 32; ++kt) {
    // S = Q K^T for 16 q rows x 32 keys
    v8f s0 = v8f{0.f,0.f,0.f,0.f,0.f,0.f,0.f,0.f};
    v8f s1 = v8f{0.f,0.f,0.f,0.f,0.f,0.f,0.f,0.f};
    for (int dc = 0; dc < 4; ++dc) {
      const bf16_t* kb0 = Kh + (size_t)(kt * 32 + lo16) * HD + dc * 32 + hi * 16;
      const bf16_t* kb1 = Kh + (size_t)(kt * 32 + 16 + lo16) * HD + dc * 32 + hi * 16;
      v16bf b0 = *(const v16bf*)kb0;
      v16bf b1 = *(const v16bf*)kb1;
      s0 = wmma_bf16(qf[dc], b0, s0);
      s1 = wmma_bf16(qf[dc], b1, s1);
    }
    // online softmax; row m = r + 8*hi lives in one 16-lane half
    float mnew[8], corr[8];
    for (int r = 0; r < 8; ++r) {
      float mx = fmaxf(s0[r] * scale, s1[r] * scale);
      for (int d = 1; d < 16; d <<= 1) mx = fmaxf(mx, __shfl_xor(mx, d, 32));
      mnew[r] = fmaxf(mrow[r], mx);
      corr[r] = __expf(mrow[r] - mnew[r]);
    }
    float p0[8], p1[8];
    for (int r = 0; r < 8; ++r) {
      p0[r] = __expf(s0[r] * scale - mnew[r]);
      p1[r] = __expf(s1[r] * scale - mnew[r]);
      float rs = p0[r] + p1[r];
      for (int d = 1; d < 16; d <<= 1) rs += __shfl_xor(rs, d, 32);
      lrow[r] = lrow[r] * corr[r] + rs;
      mrow[r] = mnew[r];
    }
    for (int dc = 0; dc < 8; ++dc)
      for (int r = 0; r < 8; ++r) o[dc][r] *= corr[r];

    // P: C-layout regs -> LDS -> A-layout frag (bf16)
    __syncthreads();
    for (int r = 0; r < 8; ++r) {
      Pls[r + 8 * hi][lo16]      = f2bf(p0[r]);
      Pls[r + 8 * hi][16 + lo16] = f2bf(p1[r]);
    }
    __syncthreads();
    v16bf pa;
    {
      v8bf a0 = *(const v8bf*)&Pls[lo16][hi * 16];
      v8bf a1 = *(const v8bf*)&Pls[lo16][hi * 16 + 8];
      pa = cat8(a0, a1);
    }
    // O += P(16x32) * V(32x128): 8 d-chunks of 16 cols, V d-major -> contiguous
    for (int dc = 0; dc < 8; ++dc) {
      const bf16_t* vb = Vh + (size_t)(dc * 16 + lo16) * LTOT + kt * 32 + hi * 16;
      v16bf b = *(const v16bf*)vb;
      o[dc] = wmma_bf16(pa, b, o[dc]);
    }
    __syncthreads();
  }

  for (int dc = 0; dc < 8; ++dc)
    for (int r = 0; r < 8; ++r) {
      int m = qt * 16 + r + 8 * hi;
      int n = dc * 16 + lo16;
      O[(size_t)m * (NH * HD) + h * HD + n] = o[dc][r] / lrow[r];
    }
}

// ---------------------------------------------------------------------------
// Elementwise / small kernels
// ---------------------------------------------------------------------------

// out[m,n] = bias[n] + sum_k A[m,k]*W[k,n]  (plain; for K=12 / N=12 / GEMV)
__global__ void simple_gemm_kernel(const float* __restrict__ A,
                                   const float* __restrict__ W,
                                   const float* __restrict__ bias,
                                   float* __restrict__ C, int M, int N, int K) {
  int idx = blockIdx.x * blockDim.x + threadIdx.x;
  if (idx >= M * N) return;
  int m = idx / N, n = idx % N;
  float acc = bias ? bias[n] : 0.f;
  for (int k = 0; k < K; ++k) acc += A[(size_t)m * K + k] * W[(size_t)k * N + n];
  C[idx] = acc;
}

// out[n] = b[n] + sum_k silu(keyv[k]) * W[k,n]
__global__ void modulation_kernel(const float* __restrict__ keyv,
                                  const float* __restrict__ W,
                                  const float* __restrict__ b,
                                  float* __restrict__ out, int N) {
  __shared__ float sk[HID];
  int t = threadIdx.x;
  for (int i = t; i < HID; i += 256) {
    float v = keyv[i];
    sk[i] = v / (1.f + __expf(-v));
  }
  __syncthreads();
  int n = blockIdx.x * 256 + t;
  if (n < N) {
    float acc = b[n];
    for (int k = 0; k < HID; ++k) acc += sk[k] * W[(size_t)k * N + n];
    out[n] = acc;
  }
}

// Y[row,:] = (1+sc)*lnorm(X[row,:]) + sh ; cols = 1024, block = 256
__global__ void lnorm_mod_kernel(const float* __restrict__ X, int ldx,
                                 const float* __restrict__ sh,
                                 const float* __restrict__ sc,
                                 float* __restrict__ Y, int ldy) {
  __shared__ float r1[256], r2[256];
  int row = blockIdx.x, t = threadIdx.x;
  const float* x = X + (size_t)row * ldx;
  float v[4]; float s = 0.f, s2 = 0.f;
  for (int i = 0; i < 4; ++i) {
    v[i] = x[t + i * 256];
    s += v[i]; s2 += v[i] * v[i];
  }
  r1[t] = s; r2[t] = s2; __syncthreads();
  for (int st = 128; st > 0; st >>= 1) {
    if (t < st) { r1[t] += r1[t + st]; r2[t] += r2[t + st]; }
    __syncthreads();
  }
  float mean = r1[0] * (1.f / 1024.f);
  float var  = r2[0] * (1.f / 1024.f) - mean * mean;
  float inv  = rsqrtf(var + 1e-6f);
  float* y = Y + (size_t)row * ldy;
  for (int i = 0; i < 4; ++i) {
    int n = t + i * 256;
    y[n] = (v[i] - mean) * inv * (1.f + sc[n]) + sh[n];
  }
}

// RMSNorm(q,k) + RoPE + pack to bf16 attention buffers.
// H: [L][ldh] with qkv in cols [0,3072): q|k|v each [NH][HD].
// grid = (L, NH), block = 128 (one thread per d)
__global__ void qkv_post_kernel(const float* __restrict__ H, int ldh, int tokOff,
                                const float* __restrict__ qn,
                                const float* __restrict__ kn,
                                bf16_t* __restrict__ Qd, bf16_t* __restrict__ Kd,
                                bf16_t* __restrict__ Vt) {
  __shared__ float red[128];
  __shared__ float sq[128], sk2[128];
  int l = blockIdx.x, h = blockIdx.y, t = threadIdx.x;
  const float* base = H + (size_t)l * ldh + h * HD;
  float qv = base[t];
  float kv = base[HID + t];
  float vv = base[2 * HID + t];

  red[t] = qv * qv; __syncthreads();
  for (int st = 64; st > 0; st >>= 1) { if (t < st) red[t] += red[t + st]; __syncthreads(); }
  float qr = rsqrtf(red[0] * (1.f / 128.f) + 1e-6f);
  __syncthreads();
  red[t] = kv * kv; __syncthreads();
  for (int st = 64; st > 0; st >>= 1) { if (t < st) red[t] += red[t + st]; __syncthreads(); }
  float kr = rsqrtf(red[0] * (1.f / 128.f) + 1e-6f);
  __syncthreads();

  sq[t]  = qv * qr * qn[t];
  sk2[t] = kv * kr * kn[t];
  __syncthreads();

  // RoPE: axes (16,56,56) -> freqs [0,8)=axis0(pos 0), [8,36)=row, [36,64)=col
  int g = tokOff + l;
  int i = t >> 1;
  float c = 1.f, s = 0.f;
  if (g >= LW && i >= 8) {
    int idx = g - LW;
    int row = idx >> 6, col = idx & 63;
    int j; float pos;
    if (i < 36) { j = i - 8;  pos = (float)row; }
    else        { j = i - 36; pos = (float)col; }
    float ang = pos * __powf(10000.f, -2.f * (float)j / 56.f);
    c = __cosf(ang); s = __sinf(ang);
  }
  float q0 = sq[i * 2],  q1 = sq[i * 2 + 1];
  float k0 = sk2[i * 2], k1 = sk2[i * 2 + 1];
  float qo = (t & 1) ? (s * q0 + c * q1) : (c * q0 - s * q1);
  float ko = (t & 1) ? (s * k0 + c * k1) : (c * k0 - s * k1);

  size_t off = (size_t)h * LTOT * HD + (size_t)g * HD + t;
  Qd[off] = f2bf(qo);
  Kd[off] = f2bf(ko);
  Vt[(size_t)h * HD * LTOT + (size_t)t * LTOT + g] = f2bf(vv);
}

// X[i] += g[i % 1024] * Y[i]
__global__ void resgate_kernel(float* __restrict__ X, const float* __restrict__ Y,
                               const float* __restrict__ g, int total) {
  int idx = blockIdx.x * blockDim.x + threadIdx.x;
  if (idx < total) X[idx] += g[idx & (HID - 1)] * Y[idx];
}

// gelu(tanh approx) in place on a strided [rows x cols] view
__global__ void gelu_kernel(float* __restrict__ P, int ld, int rows, int cols) {
  int idx = blockIdx.x * blockDim.x + threadIdx.x;
  if (idx >= rows * cols) return;
  int r = idx / cols, c = idx % cols;
  float x = P[(size_t)r * ld + c];
  float x3 = x * x * x;
  float tv = tanhf(0.7978845608028654f * (x + 0.044715f * x3));
  P[(size_t)r * ld + c] = 0.5f * x * (1.f + tv);
}

__global__ void embed_kernel(const int* __restrict__ wm,
                             const float* __restrict__ emb,
                             float* __restrict__ out) {
  int idx = blockIdx.x * blockDim.x + threadIdx.x;
  if (idx < LW * HID) {
    int l = idx >> 10, n = idx & (HID - 1);
    out[idx] = emb[(size_t)wm[l] * HID + n];
  }
}

__global__ void copy_kernel(float* __restrict__ dst, const float* __restrict__ src,
                            int n) {
  int idx = blockIdx.x * blockDim.x + threadIdx.x;
  if (idx < n) dst[idx] = src[idx];
}

// ---------------------------------------------------------------------------
// Host orchestration
// ---------------------------------------------------------------------------
static inline int cdiv(int a, int b) { return (a + b - 1) / b; }

extern "C" void kernel_launch(void* const* d_in, const int* in_sizes, int n_in,
                              void* d_out, int out_size, void* d_ws, size_t ws_size,
                              hipStream_t stream) {
  (void)in_sizes; (void)n_in; (void)out_size;

  // ---- inputs (setup_inputs dict order, params flattened in insertion order)
  const float* x_in     = (const float*)d_in[0];
  const float* key_feat = (const float*)d_in[1];
  const int*   wm_idx   = (const int*)d_in[2];
  const float* x_in_w   = (const float*)d_in[3];
  const float* x_in_b   = (const float*)d_in[4];
  const float* key_in_w = (const float*)d_in[5];
  const float* key_in_b = (const float*)d_in[6];
  const float* wm_emb   = (const float*)d_in[7];
  const int D = 8;   // double block params
  const float* x_mod_w  = (const float*)d_in[D + 0];
  const float* x_mod_b  = (const float*)d_in[D + 1];
  const float* x_qkv_w  = (const float*)d_in[D + 2];
  const float* x_qkv_b  = (const float*)d_in[D + 3];
  const float* x_qn     = (const float*)d_in[D + 4];
  const float* x_kn     = (const float*)d_in[D + 5];
  const float* x_proj_w = (const float*)d_in[D + 6];
  const float* x_proj_b = (const float*)d_in[D + 7];
  const float* x_mlp_w1 = (const float*)d_in[D + 8];
  const float* x_mlp_b1 = (const float*)d_in[D + 9];
  const float* x_mlp_w2 = (const float*)d_in[D + 10];
  const float* x_mlp_b2 = (const float*)d_in[D + 11];
  const float* w_mod_w  = (const float*)d_in[D + 12];
  const float* w_mod_b  = (const float*)d_in[D + 13];
  const float* w_qkv_w  = (const float*)d_in[D + 14];
  const float* w_qkv_b  = (const float*)d_in[D + 15];
  const float* w_qn     = (const float*)d_in[D + 16];
  const float* w_kn     = (const float*)d_in[D + 17];
  const float* w_proj_w = (const float*)d_in[D + 18];
  const float* w_proj_b = (const float*)d_in[D + 19];
  const float* w_mlp_w1 = (const float*)d_in[D + 20];
  const float* w_mlp_b1 = (const float*)d_in[D + 21];
  const float* w_mlp_w2 = (const float*)d_in[D + 22];
  const float* w_mlp_b2 = (const float*)d_in[D + 23];
  const int S0 = 32, S1 = 40, F = 48;   // single blocks, final
  const float* fin_adaln_w = (const float*)d_in[F + 0];
  const float* fin_adaln_b = (const float*)d_in[F + 1];
  const float* fin_lin_w   = (const float*)d_in[F + 2];
  const float* fin_lin_b   = (const float*)d_in[F + 3];

  float* out = (float*)d_out;

  // ---- workspace carving
  char* base = (char*)d_ws;
  size_t off = 0;
  auto carve = [&](size_t bytes) -> char* {
    char* p = base + off;
    off = (off + bytes + 255) & ~(size_t)255;
    return p;
  };
  float*  xh   = (float*)carve((size_t)LIMG * HID * 4);
  float*  wmh  = (float*)carve((size_t)LW * HID * 4);
  float*  keyv = (float*)carve((size_t)HID * 4);
  float*  modX = (float*)carve((size_t)6 * HID * 4);
  float*  modW = (float*)carve((size_t)6 * HID * 4);
  float*  xcat = (float*)carve((size_t)LTOT * HID * 4);
  float*  lnb  = (float*)carve((size_t)LTOT * HID * 4);
  float*  big  = (float*)carve((size_t)LTOT * 7168 * 4);
  float*  attn = (float*)carve((size_t)LTOT * HID * 4);
  float*  tmp2 = (float*)carve((size_t)LTOT * HID * 4);
  bf16_t* Qb   = (bf16_t*)carve((size_t)NH * LTOT * HD * 2);
  bf16_t* Kb   = (bf16_t*)carve((size_t)NH * LTOT * HD * 2);
  bf16_t* Vtb  = (bf16_t*)carve((size_t)NH * HD * LTOT * 2);
  // transposed bf16 weights
  bf16_t* xqkvT = (bf16_t*)carve((size_t)3 * HID * HID * 2);
  bf16_t* wqkvT = (bf16_t*)carve((size_t)3 * HID * HID * 2);
  bf16_t* xprjT = (bf16_t*)carve((size_t)HID * HID * 2);
  bf16_t* wprjT = (bf16_t*)carve((size_t)HID * HID * 2);
  bf16_t* xm1T  = (bf16_t*)carve((size_t)4096 * HID * 2);
  bf16_t* wm1T  = (bf16_t*)carve((size_t)4096 * HID * 2);
  bf16_t* xm2T  = (bf16_t*)carve((size_t)HID * 4096 * 2);
  bf16_t* wm2T  = (bf16_t*)carve((size_t)HID * 4096 * 2);
  bf16_t* lin1T[2], *lin2T[2];
  for (int s = 0; s < 2; ++s) {
    lin1T[s] = (bf16_t*)carve((size_t)7168 * HID * 2);
    lin2T[s] = (bf16_t*)carve((size_t)HID * 5120 * 2);
  }
  if (off > ws_size) return;  // workspace too small; nothing sane to do

  // ---- weight pre-pass: transpose + bf16
  auto cvt = [&](const float* W, bf16_t* Wt, int K, int N) {
    convert_transpose_kernel<<<dim3(N / 32, K / 32), dim3(32, 8), 0, stream>>>(
        W, Wt, K, N);
  };
  cvt(x_qkv_w, xqkvT, HID, 3 * HID);
  cvt(w_qkv_w, wqkvT, HID, 3 * HID);
  cvt(x_proj_w, xprjT, HID, HID);
  cvt(w_proj_w, wprjT, HID, HID);
  cvt(x_mlp_w1, xm1T, HID, 4096);
  cvt(w_mlp_w1, wm1T, HID, 4096);
  cvt(x_mlp_w2, xm2T, 4096, HID);
  cvt(w_mlp_w2, wm2T, 4096, HID);
  for (int s = 0; s < 2; ++s) {
    cvt((const float*)d_in[(s ? S1 : S0) + 2], lin1T[s], HID, 7168);
    cvt((const float*)d_in[(s ? S1 : S0) + 4], lin2T[s], 5120, HID);
  }

  auto gemm = [&](const float* A, int lda, const bf16_t* Wt, int ldwt,
                  const float* bias, float* C, int ldc, int M, int N, int K,
                  int acc) {
    dim3 grid(cdiv(N, 256), cdiv(M, 64));
    gemm_wmma_kernel<<<grid, dim3(256), 0, stream>>>(A, lda, Wt, ldwt, bias, C,
                                                     ldc, M, N, K, acc);
  };

  // ---- input projections
  simple_gemm_kernel<<<cdiv(LIMG * HID, 256), 256, 0, stream>>>(
      x_in, x_in_w, x_in_b, xh, LIMG, HID, 12);
  simple_gemm_kernel<<<cdiv(HID, 256), 256, 0, stream>>>(
      key_feat, key_in_w, key_in_b, keyv, 1, HID, 32);
  embed_kernel<<<cdiv(LW * HID, 256), 256, 0, stream>>>(wm_idx, wm_emb, wmh);

  // =========================== double block ===============================
  modulation_kernel<<<cdiv(6 * HID, 256), 256, 0, stream>>>(keyv, x_mod_w, x_mod_b,
                                                            modX, 6 * HID);
  modulation_kernel<<<cdiv(6 * HID, 256), 256, 0, stream>>>(keyv, w_mod_w, w_mod_b,
                                                            modW, 6 * HID);
  // x qkv
  lnorm_mod_kernel<<<LIMG, 256, 0, stream>>>(xh, HID, modX, modX + HID, lnb, HID);
  gemm(lnb, HID, xqkvT, HID, x_qkv_b, big, 3 * HID, LIMG, 3 * HID, HID, 0);
  qkv_post_kernel<<<dim3(LIMG, NH), 128, 0, stream>>>(big, 3 * HID, LW,
                                                      x_qn, x_kn, Qb, Kb, Vtb);
  // wm qkv
  lnorm_mod_kernel<<<LW, 256, 0, stream>>>(wmh, HID, modW, modW + HID, lnb, HID);
  gemm(lnb, HID, wqkvT, HID, w_qkv_b, big, 3 * HID, LW, 3 * HID, HID, 0);
  qkv_post_kernel<<<dim3(LW, NH), 128, 0, stream>>>(big, 3 * HID, 0,
                                                    w_qn, w_kn, Qb, Kb, Vtb);
  // attention over all 4160 tokens
  flash_attn_kernel<<<dim3(LTOT / 16, NH), 32, 0, stream>>>(Qb, Kb, Vtb, attn);

  // x: attn proj + residual, then MLP + residual
  gemm(attn + (size_t)LW * HID, HID, xprjT, HID, x_proj_b, tmp2, HID,
       LIMG, HID, HID, 0);
  resgate_kernel<<<cdiv(LIMG * HID, 256), 256, 0, stream>>>(xh, tmp2,
                                                            modX + 2 * HID,
                                                            LIMG * HID);
  lnorm_mod_kernel<<<LIMG, 256, 0, stream>>>(xh, HID, modX + 3 * HID,
                                             modX + 4 * HID, lnb, HID);
  gemm(lnb, HID, xm1T, HID, x_mlp_b1, big, 4096, LIMG, 4096, HID, 0);
  gelu_kernel<<<cdiv(LIMG * 4096, 256), 256, 0, stream>>>(big, 4096, LIMG, 4096);
  gemm(big, 4096, xm2T, 4096, x_mlp_b2, tmp2, HID, LIMG, HID, 4096, 0);
  resgate_kernel<<<cdiv(LIMG * HID, 256), 256, 0, stream>>>(xh, tmp2,
                                                            modX + 5 * HID,
                                                            LIMG * HID);
  // wm: attn proj + residual, then MLP + residual
  gemm(attn, HID, wprjT, HID, w_proj_b, tmp2, HID, LW, HID, HID, 0);
  resgate_kernel<<<cdiv(LW * HID, 256), 256, 0, stream>>>(wmh, tmp2,
                                                          modW + 2 * HID,
                                                          LW * HID);
  lnorm_mod_kernel<<<LW, 256, 0, stream>>>(wmh, HID, modW + 3 * HID,
                                           modW + 4 * HID, lnb, HID);
  gemm(lnb, HID, wm1T, HID, w_mlp_b1, big, 4096, LW, 4096, HID, 0);
  gelu_kernel<<<cdiv(LW * 4096, 256), 256, 0, stream>>>(big, 4096, LW, 4096);
  gemm(big, 4096, wm2T, 4096, w_mlp_b2, tmp2, HID, LW, HID, 4096, 0);
  resgate_kernel<<<cdiv(LW * HID, 256), 256, 0, stream>>>(wmh, tmp2,
                                                          modW + 5 * HID,
                                                          LW * HID);
  // concat [wm, x]
  copy_kernel<<<cdiv(LW * HID, 256), 256, 0, stream>>>(xcat, wmh, LW * HID);
  copy_kernel<<<cdiv(LIMG * HID, 256), 256, 0, stream>>>(xcat + (size_t)LW * HID,
                                                         xh, LIMG * HID);

  // =========================== single blocks ==============================
  for (int sb = 0; sb < 2; ++sb) {
    const int P = (sb == 0) ? S0 : S1;
    const float* mod_w  = (const float*)d_in[P + 0];
    const float* mod_b  = (const float*)d_in[P + 1];
    const float* lin1_b = (const float*)d_in[P + 3];
    const float* lin2_b = (const float*)d_in[P + 5];
    const float* qn     = (const float*)d_in[P + 6];
    const float* kn     = (const float*)d_in[P + 7];

    modulation_kernel<<<cdiv(3 * HID, 256), 256, 0, stream>>>(keyv, mod_w, mod_b,
                                                              modX, 3 * HID);
    lnorm_mod_kernel<<<LTOT, 256, 0, stream>>>(xcat, HID, modX, modX + HID,
                                               lnb, HID);
    gemm(lnb, HID, lin1T[sb], HID, lin1_b, big, 7168, LTOT, 7168, HID, 0);
    qkv_post_kernel<<<dim3(LTOT, NH), 128, 0, stream>>>(big, 7168, 0,
                                                        qn, kn, Qb, Kb, Vtb);
    flash_attn_kernel<<<dim3(LTOT / 16, NH), 32, 0, stream>>>(Qb, Kb, Vtb, attn);
    gelu_kernel<<<cdiv(LTOT * 4096, 256), 256, 0, stream>>>(big + 3 * HID, 7168,
                                                            LTOT, 4096);
    // out = [attn | gelu(mlp)] @ lin2 + b  (split-K: two GEMMs, 2nd accumulates)
    gemm(attn, HID, lin2T[sb], 5120, lin2_b, tmp2, HID, LTOT, HID, HID, 0);
    gemm(big + 3 * HID, 7168, lin2T[sb] + 1024, 5120, nullptr, tmp2, HID,
         LTOT, HID, 4096, 1);
    resgate_kernel<<<cdiv(LTOT * HID, 256), 256, 0, stream>>>(xcat, tmp2,
                                                              modX + 2 * HID,
                                                              LTOT * HID);
  }

  // ============================== final ===================================
  modulation_kernel<<<cdiv(2 * HID, 256), 256, 0, stream>>>(keyv, fin_adaln_w,
                                                            fin_adaln_b, modX,
                                                            2 * HID);
  lnorm_mod_kernel<<<LIMG, 256, 0, stream>>>(xcat + (size_t)LW * HID, HID,
                                             modX, modX + HID, lnb, HID);
  simple_gemm_kernel<<<cdiv(LIMG * 12, 256), 256, 0, stream>>>(
      lnb, fin_lin_w, fin_lin_b, out, LIMG, 12, HID);
}
